// GraphRelationLayer_8375186227423
// MI455X (gfx1250) — matrix operand order repaired and checked by reading, not compiled, not run on hardware
//
#include <hip/hip_runtime.h>
#include <hip/hip_bf16.h>

typedef __attribute__((ext_vector_type(16))) _Float16 v16h;
typedef __attribute__((ext_vector_type(8)))  _Float16 v8h;
typedef __attribute__((ext_vector_type(2)))  _Float16 v2h;
typedef __attribute__((ext_vector_type(8)))  float    v8f;
typedef __attribute__((ext_vector_type(4)))  float    v4f;

#define LDA 40   // f16 elems per A-tile row (32 + 8 pad); 80 B row pitch keeps 16-B alignment
#define LDT 40   // f16 elems per Bt-tile row (32 + 8 pad)

__device__ __forceinline__ float gelu_tanh(float x) {
    float x3 = x * x * x;
    return 0.5f * x * (1.0f + tanhf(0.7978845608028654f * (x + 0.044715f * x3)));
}

// ---- WMMA fragment loaders (CDNA5 ISA 7.12.2 layouts, wave32) ----
// A 16x32 f16: lane<16 -> M=lane, K {0..7}U{16..23}; lane>=16 -> M=lane-16, K {8..15}U{24..31}
// Two ds_load_b128 per lane.
__device__ __forceinline__ v16h frag_a(const _Float16* sA, int rowBase, int lane) {
    int lm = lane & 15, hi = lane >> 4;
    const v8h* p = (const v8h*)(sA + (rowBase + lm) * LDA);
    v8h x = p[hi], y = p[2 + hi];
    v16h a;
#pragma unroll
    for (int i = 0; i < 8; ++i) { a[i] = x[i]; a[i + 8] = y[i]; }
    return a;
}
// B 32x16 f16 from transposed LDS tile sBt[n][k]: N=colBase+(lane&15), K {hi*8..hi*8+7}U{16+hi*8..16+hi*8+7}.
// Two ds_load_b128 per lane.
__device__ __forceinline__ v16h frag_bt(const _Float16* sBt, int colBase, int lane) {
    int lm = lane & 15, hi = lane >> 4;
    const v8h* p = (const v8h*)(sBt + (colBase + lm) * LDT);
    v8h x = p[hi], y = p[2 + hi];
    v16h b;
#pragma unroll
    for (int i = 0; i < 8; ++i) { b[i] = x[i]; b[i + 8] = y[i]; }
    return b;
}

// ---- branchless chunked staging (one 8-f16 chunk per thread, blockDim=256) ----
__device__ __forceinline__ void stage_a_f16(_Float16* sA, const _Float16* __restrict__ src,
                                            long ld, int mBase, int M, int k0, int K, int tid) {
    int r = tid >> 2, c8 = (tid & 3) * 8;
    int gr = mBase + r, gk = k0 + c8;
    bool ok = (gr < M) && (gk + 8 <= K);
    long gro = ok ? gr : 0, gko = ok ? gk : 0;
    v8h z = {};
    v8h v = *(const v8h*)(src + gro * ld + gko);
    *(v8h*)(sA + r * LDA + c8) = ok ? v : z;
}
__device__ __forceinline__ void stage_a_f32(_Float16* sA, const float* __restrict__ src,
                                            long ld, int mBase, int M, int k0, int K, int tid) {
    int r = tid >> 2, c8 = (tid & 3) * 8;
    int gr = mBase + r, gk = k0 + c8;
    bool ok = (gr < M) && (gk + 8 <= K);
    long gro = ok ? gr : 0, gko = ok ? gk : 0;
    v4f x = *(const v4f*)(src + gro * ld + gko);
    v4f y = *(const v4f*)(src + gro * ld + gko + 4);
    v8h v;
#pragma unroll
    for (int j = 0; j < 4; ++j) { v[j] = (_Float16)x[j]; v[j + 4] = (_Float16)y[j]; }
    v8h z = {};
    *(v8h*)(sA + r * LDA + c8) = ok ? v : z;
}
// stage B (K x N row-major source) transposed into sBt[n][k]
__device__ __forceinline__ void stage_bt_f16(_Float16* sBt, const _Float16* __restrict__ src,
                                             long ldn, int nBase, int N, int k0, int K, int tid) {
    int kr = tid >> 3, c8 = (tid & 7) * 8;
    int gk = k0 + kr, gc = nBase + c8;
    bool ok = (gk < K) && (gc + 8 <= N);
    long gko = ok ? gk : 0, gco = ok ? gc : 0;
    v8h z = {};
    v8h v = *(const v8h*)(src + gko * ldn + gco);
    v = ok ? v : z;
#pragma unroll
    for (int j = 0; j < 8; ++j) sBt[(c8 + j) * LDT + kr] = v[j];
}
__device__ __forceinline__ void stage_bt_f32(_Float16* sBt, const float* __restrict__ src,
                                             long ldn, int nBase, int N, int k0, int K, int tid) {
    int kr = tid >> 3, c8 = (tid & 7) * 8;
    int gk = k0 + kr, gc = nBase + c8;
    bool ok = (gk < K) && (gc + 8 <= N);
    long gko = ok ? gk : 0, gco = ok ? gc : 0;
    v4f x = *(const v4f*)(src + gko * ldn + gco);
    v4f y = *(const v4f*)(src + gko * ldn + gco + 4);
    v8h v;
#pragma unroll
    for (int j = 0; j < 4; ++j) { v[j] = (_Float16)x[j]; v[j + 4] = (_Float16)y[j]; }
    v8h z = {};
    v = ok ? v : z;
#pragma unroll
    for (int j = 0; j < 8; ++j) sBt[(c8 + j) * LDT + kr] = v[j];
}

// =====================================================================
// 1) Attention scalars: 4 MLPs on S + exp-normalizations + type gathers
// grid = bs(32), block = 128
// =====================================================================
__global__ void attn_kernel(const float* __restrict__ S, const int* __restrict__ node_type,
                            const float* sW1, const float* sb1, const float* sW2, const float* sb2,
                            const float* eW1, const float* eb1, const float* eW2, const float* eb2,
                            const float* uW1, const float* ub1, const float* uW2, const float* ub2,
                            const float* pW1, const float* pb1, const float* pW2, const float* pb2,
                            const float* trans_scores,
                            float* start_attn, float* end_attn, float* uni, float* T) {
    __shared__ float sS[1024];
    __shared__ float sHid[128];
    __shared__ float sOut[1024];
    __shared__ float sRed[128];
    int b = blockIdx.x, tid = threadIdx.x;

    for (int i = tid; i < 1024; i += 128) sS[i] = S[b * 1024 + i];
    __syncthreads();

    auto hidden = [&](const float* W1, const float* b1) {
        float acc = b1[tid];
        for (int i = 0; i < 1024; ++i) acc += sS[i] * W1[i * 128 + tid];
        sHid[tid] = gelu_tanh(acc);
    };

    // ---- start MLP ----
    hidden(sW1, sb1); __syncthreads();
    if (tid < 3) {
        float acc = sb2[tid];
        for (int i = 0; i < 128; ++i) acc += sHid[i] * sW2[i * 3 + tid];
        sOut[tid] = acc;
    }
    __syncthreads();
    if (tid == 0) {
        float m = fmaxf(sOut[0], fmaxf(sOut[1], sOut[2]));
        sOut[4] = expf(sOut[0] - m); sOut[5] = expf(sOut[1] - m); sOut[6] = expf(sOut[2] - m);
    }
    __syncthreads();
    for (int n = tid; n < 200; n += 128)
        start_attn[b * 200 + n] = sOut[4 + node_type[b * 200 + n]];
    __syncthreads();

    // ---- end MLP ----
    hidden(eW1, eb1); __syncthreads();
    if (tid < 3) {
        float acc = eb2[tid];
        for (int i = 0; i < 128; ++i) acc += sHid[i] * eW2[i * 3 + tid];
        sOut[tid] = acc;
    }
    __syncthreads();
    if (tid == 0) {
        float m = fmaxf(sOut[0], fmaxf(sOut[1], sOut[2]));
        sOut[4] = expf(sOut[0] - m); sOut[5] = expf(sOut[1] - m); sOut[6] = expf(sOut[2] - m);
    }
    __syncthreads();
    for (int n = tid; n < 200; n += 128)
        end_attn[b * 200 + n] = sOut[4 + node_type[b * 200 + n]];
    __syncthreads();

    // ---- uni MLP (64 outputs), exp-norm over heads per k-slot ----
    hidden(uW1, ub1); __syncthreads();
    if (tid < 64) {
        float acc = ub2[tid];
        for (int i = 0; i < 128; ++i) acc += sHid[i] * uW2[i * 64 + tid];
        sOut[tid] = acc;
    }
    __syncthreads();
    if (tid < 2) {
        float m = -1e30f;
        for (int h = 0; h < 32; ++h) m = fmaxf(m, sOut[h * 2 + tid]);
        sOut[64 + tid] = m;
    }
    __syncthreads();
    if (tid < 64) uni[b * 64 + tid] = expf(sOut[tid] - sOut[64 + (tid & 1)]);
    __syncthreads();

    // ---- pair (T) MLP: 1024 outputs + trans_scores, exp-norm over all 1024 ----
    hidden(pW1, pb1); __syncthreads();
    float lmax = -1e30f;
#pragma unroll
    for (int jj = 0; jj < 8; ++jj) {
        int j = tid * 8 + jj;
        float acc = pb2[j] + trans_scores[j];
        for (int i = 0; i < 128; ++i) acc += sHid[i] * pW2[i * 1024 + j];
        sOut[j] = acc;
        lmax = fmaxf(lmax, acc);
    }
    sRed[tid] = lmax;
    __syncthreads();
    for (int s = 64; s > 0; s >>= 1) { if (tid < s) sRed[tid] = fmaxf(sRed[tid], sRed[tid + s]); __syncthreads(); }
    float mx = sRed[0];
#pragma unroll
    for (int jj = 0; jj < 8; ++jj) {
        int j = tid * 8 + jj;
        T[b * 1024 + j] = expf(sOut[j] - mx);
    }
}

// =====================================================================
// 2) Projection: Hflat(6400x200) @ U(200x600) + b, gather by node_type
// grid = (100, 10), block = 256
// =====================================================================
__global__ void __launch_bounds__(256) proj_kernel(const float* __restrict__ H, const float* __restrict__ U,
                                                   const float* __restrict__ bvec, const int* __restrict__ node_type,
                                                   const float* __restrict__ start_attn,
                                                   _Float16* __restrict__ X16, _Float16* __restrict__ Z016) {
    __shared__ _Float16 sA[64 * LDA];
    __shared__ _Float16 sBt[64 * LDT];
    int tid = threadIdx.x, wid = tid >> 5, lane = tid & 31;
    int wm = wid & 3, wn = wid >> 2;
    int mBase = blockIdx.x * 64, nBase = blockIdx.y * 64;
    v8f c0 = {}, c1 = {};
    for (int k0 = 0; k0 < 200; k0 += 32) {
        stage_a_f32(sA, H, 200, mBase, 6400, k0, 200, tid);
        stage_bt_f32(sBt, U, 600, nBase, 600, k0, 200, tid);
        __syncthreads();
        v16h a = frag_a(sA, wm * 16, lane);
        v16h b0 = frag_bt(sBt, wn * 32, lane);
        v16h b1 = frag_bt(sBt, wn * 32 + 16, lane);
        c0 = __builtin_amdgcn_wmma_f32_16x16x32_f16(false, a, false, b0, (short)0, c0, false, false);
        c1 = __builtin_amdgcn_wmma_f32_16x16x32_f16(false, a, false, b1, (short)0, c1, false, false);
        __syncthreads();
    }
    int lm = lane & 15, hi = lane >> 4;
#pragma unroll
    for (int v = 0; v < 8; ++v) {
        int r = mBase + wm * 16 + v + hi * 8;
        if (r >= 6400) continue;
        int nt = node_type[r];
        float sa = start_attn[r];
#pragma unroll
        for (int half = 0; half < 2; ++half) {
            int col = nBase + wn * 32 + lm + half * 16;
            if (col >= 600) continue;
            int jt = col / 200, cc = col - jt * 200;
            if (jt == nt) {
                float val = (half ? c1[v] : c0[v]) + bvec[col];
                X16[r * 200 + cc] = (_Float16)val;
                Z016[r * 200 + cc] = (_Float16)(val * sa);
            }
        }
    }
}

// =====================================================================
// 3) Repack W (k,h,h*n_head) f32 -> Wh (k,32,200,200) f16
// =====================================================================
__global__ void prep_W_kernel(const float* __restrict__ W, _Float16* __restrict__ Wh) {
    long total = 2L * 32 * 200 * 200;
    for (long idx = (long)blockIdx.x * 256 + threadIdx.x; idx < total; idx += (long)gridDim.x * 256) {
        long f = idx % 200, e = (idx / 200) % 200, m = (idx / 40000) % 32, t = idx / (40000L * 32);
        Wh[idx] = (_Float16)W[t * (200L * 6400) + e * 6400 + f * 32 + m];
    }
}

// =====================================================================
// 4) Per-head W GEMM: Out[b,m] = In(b[,m]) @ Wh[m] * uni[b,m,t]
// grid = (16, 32, 32), block = 256
// =====================================================================
__global__ void __launch_bounds__(256) gemm_headW_kernel(const _Float16* __restrict__ In, long sb, long sm,
                                                         const _Float16* __restrict__ Wh,
                                                         const float* __restrict__ uni, int t,
                                                         _Float16* __restrict__ Out) {
    __shared__ _Float16 sA[64 * LDA];
    __shared__ _Float16 sBt[64 * LDT];
    int tid = threadIdx.x, wid = tid >> 5, lane = tid & 31;
    int wm = wid & 3, wn = wid >> 2;
    int m = blockIdx.y, b = blockIdx.z;
    int tm = blockIdx.x & 3, tn = blockIdx.x >> 2;
    int mBase = tm * 64, nBase = tn * 64;
    const _Float16* inp = In + (long)b * sb + (long)m * sm;
    const _Float16* w = Wh + (long)m * 40000;
    float scale = uni[(b * 32 + m) * 2 + t];
    v8f c0 = {}, c1 = {};
    for (int k0 = 0; k0 < 200; k0 += 32) {
        stage_a_f16(sA, inp, 200, mBase, 200, k0, 200, tid);
        stage_bt_f16(sBt, w, 200, nBase, 200, k0, 200, tid);
        __syncthreads();
        v16h a = frag_a(sA, wm * 16, lane);
        v16h b0 = frag_bt(sBt, wn * 32, lane);
        v16h b1 = frag_bt(sBt, wn * 32 + 16, lane);
        c0 = __builtin_amdgcn_wmma_f32_16x16x32_f16(false, a, false, b0, (short)0, c0, false, false);
        c1 = __builtin_amdgcn_wmma_f32_16x16x32_f16(false, a, false, b1, (short)0, c1, false, false);
        __syncthreads();
    }
    _Float16* op = Out + (long)(b * 32 + m) * 40000;
    int lm = lane & 15, hi = lane >> 4;
#pragma unroll
    for (int v = 0; v < 8; ++v) {
        int r = mBase + wm * 16 + v + hi * 8;
        if (r >= 200) continue;
        int c = nBase + wn * 32 + lm;
        if (c < 200)      op[r * 200 + c]      = (_Float16)(c0[v] * scale);
        if (c + 16 < 200) op[r * 200 + c + 16] = (_Float16)(c1[v] * scale);
    }
}

// =====================================================================
// 5) Per-head adjacency aggregation: Out[b,m] = A[b,m] @ Zin[b,m]
// grid = (16, 32, 32), block = 256
// =====================================================================
__global__ void __launch_bounds__(256) agg_A_kernel(const float* __restrict__ A,
                                                    const _Float16* __restrict__ Zin,
                                                    _Float16* __restrict__ Zout) {
    __shared__ _Float16 sA[64 * LDA];
    __shared__ _Float16 sBt[64 * LDT];
    int tid = threadIdx.x, wid = tid >> 5, lane = tid & 31;
    int wm = wid & 3, wn = wid >> 2;
    int m = blockIdx.y, b = blockIdx.z;
    int tm = blockIdx.x & 3, tn = blockIdx.x >> 2;
    int mBase = tm * 64, nBase = tn * 64;
    const float* Ap = A + (long)(b * 32 + m) * 40000;
    const _Float16* zp = Zin + (long)(b * 32 + m) * 40000;
    v8f c0 = {}, c1 = {};
    for (int k0 = 0; k0 < 200; k0 += 32) {
        stage_a_f32(sA, Ap, 200, mBase, 200, k0, 200, tid);
        stage_bt_f16(sBt, zp, 200, nBase, 200, k0, 200, tid);
        __syncthreads();
        v16h a = frag_a(sA, wm * 16, lane);
        v16h b0 = frag_bt(sBt, wn * 32, lane);
        v16h b1 = frag_bt(sBt, wn * 32 + 16, lane);
        c0 = __builtin_amdgcn_wmma_f32_16x16x32_f16(false, a, false, b0, (short)0, c0, false, false);
        c1 = __builtin_amdgcn_wmma_f32_16x16x32_f16(false, a, false, b1, (short)0, c1, false, false);
        __syncthreads();
    }
    _Float16* op = Zout + (long)(b * 32 + m) * 40000;
    int lm = lane & 15, hi = lane >> 4;
#pragma unroll
    for (int v = 0; v < 8; ++v) {
        int r = mBase + wm * 16 + v + hi * 8;
        if (r >= 200) continue;
        int c = nBase + wn * 32 + lm;
        if (c < 200)      op[r * 200 + c]      = (_Float16)c0[v];
        if (c + 16 < 200) op[r * 200 + c + 16] = (_Float16)c1[v];
    }
}

// =====================================================================
// 6) Head-transition mix: Z3[b,m,ne] = sum_h T[b,h,m] * Z2[b,h,ne]
// 2-wide packed f16; grid = (79, 32), block = 256
// =====================================================================
__global__ void tmix_kernel(const _Float16* __restrict__ Z2, const float* __restrict__ T,
                            _Float16* __restrict__ Z3) {
    __shared__ float sT[1024];
    int b = blockIdx.y, tid = threadIdx.x;
    for (int i = tid; i < 1024; i += 256) sT[i] = T[b * 1024 + i];
    __syncthreads();
    int ne = (blockIdx.x * 256 + tid) * 2;
    if (ne < 40000) {
        const v2h* zp = (const v2h*)(Z2 + (long)b * 32 * 40000 + ne);
        __builtin_prefetch(zp + 256, 0, 1);
        float z0[32], z1[32];
#pragma unroll
        for (int h = 0; h < 32; ++h) {
            v2h v = zp[(long)h * 20000];
            z0[h] = (float)v[0]; z1[h] = (float)v[1];
        }
        v2h* op = (v2h*)(Z3 + (long)b * 32 * 40000 + ne);
        for (int m = 0; m < 32; ++m) {
            float a0 = 0.f, a1 = 0.f;
#pragma unroll
            for (int h = 0; h < 32; ++h) {
                float t = sT[h * 32 + m];
                a0 += z0[h] * t; a1 += z1[h] * t;
            }
            v2h o; o[0] = (_Float16)a0; o[1] = (_Float16)a1;
            op[(long)m * 20000] = o;
        }
    }
}

// =====================================================================
// 7) Final aggregation + head sum: Zsum[b] = sum_m A[b,m] @ Z4[b,m]
// grid = (16, 32), block = 256
// =====================================================================
__global__ void __launch_bounds__(256) final_agg_kernel(const float* __restrict__ A,
                                                        const _Float16* __restrict__ Z4,
                                                        float* __restrict__ Zsum) {
    __shared__ _Float16 sA[64 * LDA];
    __shared__ _Float16 sBt[64 * LDT];
    int tid = threadIdx.x, wid = tid >> 5, lane = tid & 31;
    int wm = wid & 3, wn = wid >> 2;
    int b = blockIdx.y;
    int tm = blockIdx.x & 3, tn = blockIdx.x >> 2;
    int mBase = tm * 64, nBase = tn * 64;
    v8f c0 = {}, c1 = {};
    for (int m = 0; m < 32; ++m) {
        const float* Ap = A + (long)(b * 32 + m) * 40000;
        const _Float16* zp = Z4 + (long)(b * 32 + m) * 40000;
        for (int k0 = 0; k0 < 200; k0 += 32) {
            stage_a_f32(sA, Ap, 200, mBase, 200, k0, 200, tid);
            stage_bt_f16(sBt, zp, 200, nBase, 200, k0, 200, tid);
            __syncthreads();
            v16h a = frag_a(sA, wm * 16, lane);
            v16h b0 = frag_bt(sBt, wn * 32, lane);
            v16h b1 = frag_bt(sBt, wn * 32 + 16, lane);
            c0 = __builtin_amdgcn_wmma_f32_16x16x32_f16(false, a, false, b0, (short)0, c0, false, false);
            c1 = __builtin_amdgcn_wmma_f32_16x16x32_f16(false, a, false, b1, (short)0, c1, false, false);
            __syncthreads();
        }
    }
    int lm = lane & 15, hi = lane >> 4;
#pragma unroll
    for (int v = 0; v < 8; ++v) {
        int r = mBase + wm * 16 + v + hi * 8;
        if (r >= 200) continue;
        int c = nBase + wn * 32 + lm;
        if (c < 200)      Zsum[((long)b * 200 + r) * 200 + c]      = c0[v];
        if (c + 16 < 200) Zsum[((long)b * 200 + r) * 200 + c + 16] = c1[v];
    }
}

// =====================================================================
// 8) Denominator passes (small matvecs)
// =====================================================================
__global__ void d_pass1_kernel(const float* __restrict__ A, const float* __restrict__ start_attn,
                               const float* __restrict__ uni, float* __restrict__ D1) {
    int m = blockIdx.x, b = blockIdx.y, tid = threadIdx.x;
    __shared__ float sV[200];
    float u0 = uni[(b * 32 + m) * 2 + 0];
    for (int j = tid; j < 200; j += 256) sV[j] = start_attn[b * 200 + j] * u0;
    __syncthreads();
    int wid = tid >> 5, lane = tid & 31;
    const float* Ap = A + (long)(b * 32 + m) * 40000;
    for (int i = wid; i < 200; i += 8) {
        float acc = 0.f;
        for (int j = lane; j < 200; j += 32) acc += Ap[i * 200 + j] * sV[j];
        for (int off = 16; off > 0; off >>= 1) acc += __shfl_down(acc, off, 32);
        if (lane == 0) D1[(long)(b * 32 + m) * 200 + i] = acc;
    }
}

__global__ void d_pass2_kernel(const float* __restrict__ A, const float* __restrict__ D1,
                               const float* __restrict__ T, const float* __restrict__ uni,
                               float* __restrict__ D2) {
    int m = blockIdx.x, b = blockIdx.y, tid = threadIdx.x;
    __shared__ float sTi[32];
    __shared__ float sDm[200];
    if (tid < 32) sTi[tid] = T[b * 1024 + tid * 32 + m];
    __syncthreads();
    float u1 = uni[(b * 32 + m) * 2 + 1];
    for (int j = tid; j < 200; j += 256) {
        float acc = 0.f;
        for (int i = 0; i < 32; ++i) acc += D1[(long)(b * 32 + i) * 200 + j] * sTi[i];
        sDm[j] = acc * u1;
    }
    __syncthreads();
    int wid = tid >> 5, lane = tid & 31;
    const float* Ap = A + (long)(b * 32 + m) * 40000;
    for (int i = wid; i < 200; i += 8) {
        float acc = 0.f;
        for (int j = lane; j < 200; j += 32) acc += Ap[i * 200 + j] * sDm[j];
        for (int off = 16; off > 0; off >>= 1) acc += __shfl_down(acc, off, 32);
        if (lane == 0) D2[(long)(b * 32 + m) * 200 + i] = acc;
    }
}

__global__ void d_reduce_kernel(const float* __restrict__ D2, const float* __restrict__ end_attn,
                                float* __restrict__ Dsum) {
    int b = blockIdx.x, tid = threadIdx.x;
    for (int n = tid; n < 200; n += 256) {
        float acc = 0.f;
        for (int m = 0; m < 32; ++m) acc += D2[(long)(b * 32 + m) * 200 + n];
        Dsum[b * 200 + n] = end_attn[b * 200 + n] * acc;
    }
}

// =====================================================================
// 9) Output: gelu( X@V1 + Znorm@V2 ), Znorm = Zsum*end/(Dsum+EPS)
// grid = (100, 4), block = 256
// =====================================================================
__global__ void __launch_bounds__(256) out_kernel(const _Float16* __restrict__ X16,
                                                  const float* __restrict__ Zsum,
                                                  const float* __restrict__ end_attn,
                                                  const float* __restrict__ Dsum,
                                                  const float* __restrict__ V1, const float* __restrict__ V2,
                                                  float* __restrict__ out) {
    __shared__ _Float16 sA1[64 * LDA];
    __shared__ _Float16 sA2[64 * LDA];
    __shared__ _Float16 sB1[64 * LDT];
    __shared__ _Float16 sB2[64 * LDT];
    int tid = threadIdx.x, wid = tid >> 5, lane = tid & 31;
    int wm = wid & 3, wn = wid >> 2;
    int mBase = blockIdx.x * 64, nBase = blockIdx.y * 64;
    v8f c0 = {}, c1 = {};
    for (int k0 = 0; k0 < 200; k0 += 32) {
        stage_a_f16(sA1, X16, 200, mBase, 6400, k0, 200, tid);
        // A2 = Zsum * end/(D+eps), converted f16, same chunking (row-uniform scale)
        {
            int r = tid >> 2, c8 = (tid & 3) * 8;
            int gr = mBase + r, gk = k0 + c8;
            bool ok = (gr < 6400) && (gk + 8 <= 200);
            long gro = ok ? gr : 0, gko = ok ? gk : 0;
            float sc = end_attn[gro] / (Dsum[gro] + 1e-20f);
            v4f x = *(const v4f*)(Zsum + gro * 200 + gko);
            v4f y = *(const v4f*)(Zsum + gro * 200 + gko + 4);
            v8h v;
#pragma unroll
            for (int j = 0; j < 4; ++j) { v[j] = (_Float16)(x[j] * sc); v[j + 4] = (_Float16)(y[j] * sc); }
            v8h z = {};
            *(v8h*)(sA2 + r * LDA + c8) = ok ? v : z;
        }
        stage_bt_f32(sB1, V1, 200, nBase, 200, k0, 200, tid);
        stage_bt_f32(sB2, V2, 200, nBase, 200, k0, 200, tid);
        __syncthreads();
        v16h a1 = frag_a(sA1, wm * 16, lane);
        v16h a2 = frag_a(sA2, wm * 16, lane);
        v16h b10 = frag_bt(sB1, wn * 32, lane);
        v16h b11 = frag_bt(sB1, wn * 32 + 16, lane);
        v16h b20 = frag_bt(sB2, wn * 32, lane);
        v16h b21 = frag_bt(sB2, wn * 32 + 16, lane);
        c0 = __builtin_amdgcn_wmma_f32_16x16x32_f16(false, a1, false, b10, (short)0, c0, false, false);
        c0 = __builtin_amdgcn_wmma_f32_16x16x32_f16(false, a2, false, b20, (short)0, c0, false, false);
        c1 = __builtin_amdgcn_wmma_f32_16x16x32_f16(false, a1, false, b11, (short)0, c1, false, false);
        c1 = __builtin_amdgcn_wmma_f32_16x16x32_f16(false, a2, false, b21, (short)0, c1, false, false);
        __syncthreads();
    }
    int lm = lane & 15, hi = lane >> 4;
#pragma unroll
    for (int v = 0; v < 8; ++v) {
        int r = mBase + wm * 16 + v + hi * 8;
        if (r >= 6400) continue;
        int c = nBase + wn * 32 + lm;
        if (c < 200)      out[(long)r * 200 + c]      = gelu_tanh(c0[v]);
        if (c + 16 < 200) out[(long)r * 200 + c + 16] = gelu_tanh(c1[v]);
    }
}

// =====================================================================
extern "C" void kernel_launch(void* const* d_in, const int* in_sizes, int n_in,
                              void* d_out, int out_size, void* d_ws, size_t ws_size,
                              hipStream_t stream) {
    const float* S          = (const float*)d_in[0];
    const float* H          = (const float*)d_in[1];
    const float* A          = (const float*)d_in[2];
    const int*   node_type  = (const int*)d_in[3];
    const float* U          = (const float*)d_in[4];
    const float* bvec       = (const float*)d_in[5];
    const float* W          = (const float*)d_in[6];
    const float* V1         = (const float*)d_in[7];
    const float* V2         = (const float*)d_in[8];
    const float* sW1 = (const float*)d_in[9],  *sb1 = (const float*)d_in[10];
    const float* sW2 = (const float*)d_in[11], *sb2 = (const float*)d_in[12];
    const float* eW1 = (const float*)d_in[13], *eb1 = (const float*)d_in[14];
    const float* eW2 = (const float*)d_in[15], *eb2 = (const float*)d_in[16];
    const float* uW1 = (const float*)d_in[17], *ub1 = (const float*)d_in[18];
    const float* uW2 = (const float*)d_in[19], *ub2 = (const float*)d_in[20];
    const float* pW1 = (const float*)d_in[21], *pb1 = (const float*)d_in[22];
    const float* pW2 = (const float*)d_in[23], *pb2 = (const float*)d_in[24];
    const float* trans = (const float*)d_in[25];

    char* ws = (char*)d_ws;
    size_t cur = 0;
    auto alloc = [&](size_t bytes) -> char* {
        char* p = ws + cur;
        cur += (bytes + 255) & ~(size_t)255;
        return p;
    };

    float*     start_attn = (float*)alloc(6400 * 4);
    float*     end_attn   = (float*)alloc(6400 * 4);
    float*     uni        = (float*)alloc(2048 * 4);
    float*     T          = (float*)alloc(32768 * 4);
    _Float16*  X16        = (_Float16*)alloc(1280000L * 2);
    _Float16*  Z016       = (_Float16*)alloc(1280000L * 2);
    _Float16*  Wh16       = (_Float16*)alloc(2560000L * 2);
    _Float16*  Zb1        = (_Float16*)alloc(40960000L * 2);
    _Float16*  Zb2        = (_Float16*)alloc(40960000L * 2);
    float*     Zsum       = (float*)alloc(1280000L * 4);
    float*     D1         = (float*)alloc(204800L * 4);
    float*     D2         = (float*)alloc(204800L * 4);
    float*     Dsum       = (float*)alloc(6400 * 4);

    attn_kernel<<<32, 128, 0, stream>>>(S, node_type, sW1, sb1, sW2, sb2, eW1, eb1, eW2, eb2,
                                        uW1, ub1, uW2, ub2, pW1, pb1, pW2, pb2, trans,
                                        start_attn, end_attn, uni, T);
    proj_kernel<<<dim3(100, 10), 256, 0, stream>>>(H, U, bvec, node_type, start_attn, X16, Z016);
    prep_W_kernel<<<4096, 256, 0, stream>>>(W, Wh16);
    gemm_headW_kernel<<<dim3(16, 32, 32), 256, 0, stream>>>(Z016, 40000L, 0L, Wh16, uni, 0, Zb1);
    agg_A_kernel<<<dim3(16, 32, 32), 256, 0, stream>>>(A, Zb1, Zb2);
    tmix_kernel<<<dim3(79, 32), 256, 0, stream>>>(Zb2, T, Zb1);
    gemm_headW_kernel<<<dim3(16, 32, 32), 256, 0, stream>>>(Zb1, 32L * 40000L, 40000L,
                                                            Wh16 + 32L * 40000L, uni, 1, Zb2);
    final_agg_kernel<<<dim3(16, 32), 256, 0, stream>>>(A, Zb2, Zsum);
    d_pass1_kernel<<<dim3(32, 32), 256, 0, stream>>>(A, start_attn, uni, D1);
    d_pass2_kernel<<<dim3(32, 32), 256, 0, stream>>>(A, D1, T, uni, D2);
    d_reduce_kernel<<<32, 256, 0, stream>>>(D2, end_attn, Dsum);
    out_kernel<<<dim3(100, 4), 256, 0, stream>>>(X16, Zsum, end_attn, Dsum, V1, V2, (float*)d_out);
}
